// BiGAT_RNN_10960756539875
// MI455X (gfx1250) — matrix operand order repaired and checked by reading, not compiled, not run on hardware
//
#include <hip/hip_runtime.h>
#include <hip/hip_bf16.h>
#include <math.h>

// ---------------- sizes ----------------
#define Bq   64
#define Lq   256
#define Eq   300
#define Hq   128
#define G4H  512           // 4*H
#define TWOH 256           // 2*H
#define NH1  8
#define F1   8
#define LRELU 0.2f

typedef __attribute__((ext_vector_type(16))) _Float16 v16h;
typedef __attribute__((ext_vector_type(8)))  _Float16 v8h;
typedef __attribute__((ext_vector_type(4)))  _Float16 v4h;
typedef __attribute__((ext_vector_type(8)))  float    v8f;

// ---------------- WMMA fragment helpers (CDNA5 16x16x32 f16, wave32) --------
// A (16-bit 16x32 MxK), lane L: m = L%16, half = L/16.
//   element e: k = (e<8) ? 8*half+e : 16 + 8*half + (e-8)
//   => two contiguous 8-f16 runs at kbase+8*half and kbase+16+8*half (2x ds_load_b128)
__device__ __forceinline__ v16h load_a_frag(const _Float16* __restrict__ As,
                                            int stride, int mbase, int kbase, int lane) {
  const int half = lane >> 4;
  const _Float16* p = As + (mbase + (lane & 15)) * stride + kbase + 8 * half;
  v8h lo = *(const v8h*)p;          // K = 8*half + 0..7
  v8h hi = *(const v8h*)(p + 16);   // K = 16 + 8*half + 0..7
  return __builtin_shufflevector(lo, hi, 0,1,2,3,4,5,6,7,8,9,10,11,12,13,14,15);
}

// B (16-bit 32x16 KxN), lane L: n = L%16, half = L/16; element e -> k = (e<8) ? kbase+8*half+e : kbase+16+8*half+(e-8).
// BsT stored TRANSPOSED as [n][k] => two contiguous 8-f16 runs (2x ds_load_b128)
__device__ __forceinline__ v16h load_bT_frag(const _Float16* __restrict__ BsT,
                                             int stride, int kbase, int nbase, int lane) {
  const int half = lane >> 4;
  const _Float16* p = BsT + (nbase + (lane & 15)) * stride + kbase + 8 * half;
  v8h lo = *(const v8h*)p;
  v8h hi = *(const v8h*)(p + 16);
  return __builtin_shufflevector(lo, hi, 0,1,2,3,4,5,6,7,8,9,10,11,12,13,14,15);
}

__device__ __forceinline__ v8h cvt8(const float4 a, const float4 b) {
  v8h h;
  h[0] = (_Float16)a.x; h[1] = (_Float16)a.y; h[2] = (_Float16)a.z; h[3] = (_Float16)a.w;
  h[4] = (_Float16)b.x; h[5] = (_Float16)b.y; h[6] = (_Float16)b.z; h[7] = (_Float16)b.w;
  return h;
}

#define WMMA_F16(a, b, c) \
  __builtin_amdgcn_wmma_f32_16x16x32_f16(false, (a), false, (b), (short)0, (c), false, false)

// C/D (32-bit 16x16): VGPR r: row = r + 8*half, col = lane%16

// =====================================================================
// Kernel 1: embedding gather + input projection GEMM (both directions)
//   xproj[m][n] = sum_e emb[ids[m]][e] * W[n][e] + bias[n]
//   m in [0,16384), n in [0,1024): n<512 -> Wih_f / b_f, else Wih_b / b_b
//   K = 300 padded to 320; double-buffered LDS (global loads overlap WMMA).
// =====================================================================
#define TS 40   // LDS tile k-stride in f16 (32 + 8 pad; 20 dwords -> conflict-free)
__global__ __launch_bounds__(256) void
k_embed_xproj(const int* __restrict__ ids, const float* __restrict__ emb,
              const float* __restrict__ Wf, const float* __restrict__ Wb,
              const float* __restrict__ bf, const float* __restrict__ bb,
              float* __restrict__ xproj) {
  __shared__ alignas(16) _Float16 As [2][64 * TS];   // [m][k]
  __shared__ alignas(16) _Float16 BsT[2][64 * TS];   // [n][k] (transposed B)
  const int tid  = threadIdx.x;                   // 256 threads, 8 waves
  const int lane = tid & 31, wave = tid >> 5;
  const int m0 = blockIdx.x * 64;
  const int n0 = blockIdx.y * 64;
  const int tm0 = wave & 3, tn0 = wave >> 2;        // tile 0
  const int tm1 = wave & 3, tn1 = (wave >> 2) + 2;  // tile 1
  const int rrow = tid >> 2;                        // 0..63: A row mm == B row nn
  const int kk0  = (tid & 3) * 8;                   // 0/8/16/24

  const int myid = ids[m0 + rrow];                  // token id cached once
  const float* __restrict__ arow = emb + (long)myid * Eq;
  const int nW = n0 + rrow;
  const float* __restrict__ brow =
      (nW < G4H) ? (Wf + (long)nW * Eq) : (Wb + (long)(nW - G4H) * Eq);

  // ---- prologue: fetch slab 0 ----
  v8h av, bv;
  {
    const int k0 = kk0;
    float4 a0 = *(const float4*)(arow + k0);
    float4 a1 = *(const float4*)(arow + k0 + 4);
    float4 b0 = *(const float4*)(brow + k0);
    float4 b1 = *(const float4*)(brow + k0 + 4);
    av = cvt8(a0, a1); bv = cvt8(b0, b1);
  }
  *(v8h*)(As [0] + rrow * TS + kk0) = av;
  *(v8h*)(BsT[0] + rrow * TS + kk0) = bv;
  __syncthreads();

  v8f acc0 = {}, acc1 = {};
  for (int i = 0; i < 10; ++i) {                    // 10 slabs of K=32
    const int cur = i & 1;
    v8h nav, nbv;
    if (i < 9) {                                    // prefetch slab i+1
      const int kb = (i + 1) * 32;
      const int k0 = kb + kk0;
      if (kb <= 256) {                              // fully in-bounds: b128 loads
        float4 a0 = *(const float4*)(arow + k0);
        float4 a1 = *(const float4*)(arow + k0 + 4);
        float4 b0 = *(const float4*)(brow + k0);
        float4 b1 = *(const float4*)(brow + k0 + 4);
        nav = cvt8(a0, a1); nbv = cvt8(b0, b1);
      } else {                                      // kb == 288: clamped, branchless
#pragma unroll
        for (int q = 0; q < 8; ++q) {
          const int k = k0 + q;
          const int kc = (k < Eq) ? k : (Eq - 1);
          const float va = arow[kc], vb = brow[kc];
          nav[q] = (_Float16)((k < Eq) ? va : 0.f);
          nbv[q] = (_Float16)((k < Eq) ? vb : 0.f);
        }
      }
    }
    // compute on current buffer
    {
      v16h a = load_a_frag (As [cur], TS, tm0 * 16, 0, lane);
      v16h b = load_bT_frag(BsT[cur], TS, 0, tn0 * 16, lane);
      acc0 = WMMA_F16(a, b, acc0);
      v16h a2 = load_a_frag (As [cur], TS, tm1 * 16, 0, lane);
      v16h b2 = load_bT_frag(BsT[cur], TS, 0, tn1 * 16, lane);
      acc1 = WMMA_F16(a2, b2, acc1);
    }
    if (i < 9) {                                    // stage slab i+1 into other buffer
      *(v8h*)(As [cur ^ 1] + rrow * TS + kk0) = nav;
      *(v8h*)(BsT[cur ^ 1] + rrow * TS + kk0) = nbv;
    }
    __syncthreads();
  }
  const int half = lane >> 4;
  {
    const int n = n0 + tn0 * 16 + (lane & 15);
    const float bias = (n < G4H) ? bf[n] : bb[n - G4H];
#pragma unroll
    for (int r = 0; r < 8; ++r)
      xproj[(m0 + tm0 * 16 + 8 * half + r) * 1024 + n] = acc0[r] + bias;
  }
  {
    const int n = n0 + tn1 * 16 + (lane & 15);
    const float bias = (n < G4H) ? bf[n] : bb[n - G4H];
#pragma unroll
    for (int r = 0; r < 8; ++r)
      xproj[(m0 + tm1 * 16 + 8 * half + r) * 1024 + n] = acc1[r] + bias;
  }
}

// =====================================================================
// Kernel 2: persistent LSTM recurrence. One workgroup per direction.
// LDS: UtT f16 [512][136] (139,264B) | g f32 [64][512] (131,072B)
//      | hc f16 [64][136] (17,408B)  | c f32 [64][128] (32,768B) = 320,512B (WGP LDS)
// UtT[j][k] = Whh[j][k] row-major => B-fragment contiguous; stride 136 f16
// (68 dwords) puts the 16 fragment rows on distinct LDS banks.
// xproj for the step is prefetched into VGPRs BEFORE the WMMA loop so VMEM
// latency hides behind the 16 WMMAs + LDS traffic.
// =====================================================================
#define UT_S 136
#define HC_S 136
#define SMEM_UT   0
#define SMEM_G    139264
#define SMEM_HC   270336
#define SMEM_C    287744
#define SMEM_TOT  320512
__global__ __launch_bounds__(1024) void
k_lstm(const float* __restrict__ Whh_f, const float* __restrict__ Whh_b,
       const float* __restrict__ xproj, float* __restrict__ h_seq) {
  extern __shared__ char smem[];
  _Float16* UtT = (_Float16*)(smem + SMEM_UT);
  float*    g   = (float*)(smem + SMEM_G);
  _Float16* hc  = (_Float16*)(smem + SMEM_HC);
  float*    cs  = (float*)(smem + SMEM_C);
  const int dir = blockIdx.x;
  const float* __restrict__ U = dir ? Whh_b : Whh_f;
  const int tid = threadIdx.x;                  // 1024 threads = 32 waves
  const int lane = tid & 31, wave = tid >> 5;

  for (int e4 = tid; e4 < (G4H * Hq) / 4; e4 += 1024) {  // U -> f16 LDS (b128 loads)
    const int j  = e4 >> 5;
    const int k4 = (e4 & 31) * 4;
    float4 u = *(const float4*)(U + j * Hq + k4);
    v4h h4;
    h4[0] = (_Float16)u.x; h4[1] = (_Float16)u.y;
    h4[2] = (_Float16)u.z; h4[3] = (_Float16)u.w;
    *(v4h*)(UtT + j * UT_S + k4) = h4;
  }
  for (int e = tid; e < Bq * Hq; e += 1024) {
    hc[(e >> 7) * HC_S + (e & 127)] = (_Float16)0.f;
    cs[e] = 0.f;
  }
  __syncthreads();

  const int tn = wave;                          // N-tile 0..31
  const int half = lane >> 4;
  const int n = tn * 16 + (lane & 15);
  const int col = dir * G4H + n;

  for (int t = 0; t < Lq; ++t) {
    const int tt = dir ? (Lq - 1 - t) : t;
    // prefetch this step's xproj contributions (independent of LDS state)
    float xp0[8], xp1[8], xp2[8], xp3[8];
#pragma unroll
    for (int r = 0; r < 8; ++r) {
      const int b0 = 8 * half + r;
      xp0[r] = xproj[((b0     ) * Lq + tt) * 1024 + col];
      xp1[r] = xproj[((b0 + 16) * Lq + tt) * 1024 + col];
      xp2[r] = xproj[((b0 + 32) * Lq + tt) * 1024 + col];
      xp3[r] = xproj[((b0 + 48) * Lq + tt) * 1024 + col];
    }
    v8f a0 = {}, a1 = {}, a2 = {}, a3 = {};
#pragma unroll
    for (int kt = 0; kt < 4; ++kt) {            // K = 128 = 4 * 32
      v16h bf_ = load_bT_frag(UtT, UT_S, kt * 32, tn * 16, lane);
      v16h f0 = load_a_frag(hc, HC_S, 0,  kt * 32, lane);
      a0 = WMMA_F16(f0, bf_, a0);
      v16h f1 = load_a_frag(hc, HC_S, 16, kt * 32, lane);
      a1 = WMMA_F16(f1, bf_, a1);
      v16h f2 = load_a_frag(hc, HC_S, 32, kt * 32, lane);
      a2 = WMMA_F16(f2, bf_, a2);
      v16h f3 = load_a_frag(hc, HC_S, 48, kt * 32, lane);
      a3 = WMMA_F16(f3, bf_, a3);
    }
#pragma unroll
    for (int r = 0; r < 8; ++r) {
      const int b0 = 8 * half + r;
      g[(b0     ) * G4H + n] = a0[r] + xp0[r];
      g[(b0 + 16) * G4H + n] = a1[r] + xp1[r];
      g[(b0 + 32) * G4H + n] = a2[r] + xp2[r];
      g[(b0 + 48) * G4H + n] = a3[r] + xp3[r];
    }
    __syncthreads();
    for (int e = tid; e < Bq * Hq; e += 1024) { // gates, PyTorch order i,f,g,o
      const int bb = e >> 7, k = e & 127;
      const float gi = g[bb * G4H +          k];
      const float gf = g[bb * G4H + Hq     + k];
      const float gg = g[bb * G4H + 2 * Hq + k];
      const float go = g[bb * G4H + 3 * Hq + k];
      const float si = 1.f / (1.f + expf(-gi));
      const float sf = 1.f / (1.f + expf(-gf));
      const float so = 1.f / (1.f + expf(-go));
      float c = sf * cs[e] + si * tanhf(gg);
      const float h = so * tanhf(c);
      cs[e] = c;
      hc[bb * HC_S + k] = (_Float16)h;
      h_seq[(bb * Lq + tt) * TWOH + dir * Hq + k] = h;
    }
    __syncthreads();
  }
}

// =====================================================================
// Kernel 3: GAT1 projection GEMM: proj[16384][64] = h_seq[16384][256] @ g1_W.T
// Double-buffered LDS, all K in-bounds.
// =====================================================================
__global__ __launch_bounds__(256) void
k_gat1_proj(const float* __restrict__ hseq, const float* __restrict__ g1W,
            float* __restrict__ proj) {
  __shared__ alignas(16) _Float16 As [2][64 * TS];
  __shared__ alignas(16) _Float16 BsT[2][64 * TS];
  const int tid = threadIdx.x;              // 256 threads, 8 waves
  const int lane = tid & 31, wave = tid >> 5;
  const int m0 = blockIdx.x * 64;
  const int tm0 = wave & 3, tn0 = wave >> 2;
  const int tm1 = wave & 3, tn1 = (wave >> 2) + 2;
  const int rrow = tid >> 2;
  const int kk0  = (tid & 3) * 8;
  const float* __restrict__ arow = hseq + (long)(m0 + rrow) * TWOH;
  const float* __restrict__ brow = g1W + (long)rrow * TWOH;   // n == rrow (N=64)

  {
    float4 a0 = *(const float4*)(arow + kk0);
    float4 a1 = *(const float4*)(arow + kk0 + 4);
    float4 b0 = *(const float4*)(brow + kk0);
    float4 b1 = *(const float4*)(brow + kk0 + 4);
    *(v8h*)(As [0] + rrow * TS + kk0) = cvt8(a0, a1);
    *(v8h*)(BsT[0] + rrow * TS + kk0) = cvt8(b0, b1);
  }
  __syncthreads();

  v8f acc0 = {}, acc1 = {};
  for (int i = 0; i < 8; ++i) {                 // 8 slabs of K=32
    const int cur = i & 1;
    v8h nav, nbv;
    if (i < 7) {
      const int k0 = (i + 1) * 32 + kk0;
      float4 a0 = *(const float4*)(arow + k0);
      float4 a1 = *(const float4*)(arow + k0 + 4);
      float4 b0 = *(const float4*)(brow + k0);
      float4 b1 = *(const float4*)(brow + k0 + 4);
      nav = cvt8(a0, a1); nbv = cvt8(b0, b1);
    }
    v16h a = load_a_frag (As [cur], TS, tm0 * 16, 0, lane);
    v16h b = load_bT_frag(BsT[cur], TS, 0, tn0 * 16, lane);
    acc0 = WMMA_F16(a, b, acc0);
    v16h a2 = load_a_frag (As [cur], TS, tm1 * 16, 0, lane);
    v16h b2 = load_bT_frag(BsT[cur], TS, 0, tn1 * 16, lane);
    acc1 = WMMA_F16(a2, b2, acc1);
    if (i < 7) {
      *(v8h*)(As [cur ^ 1] + rrow * TS + kk0) = nav;
      *(v8h*)(BsT[cur ^ 1] + rrow * TS + kk0) = nbv;
    }
    __syncthreads();
  }
  const int half = lane >> 4;
#pragma unroll
  for (int r = 0; r < 8; ++r) {
    proj[(m0 + tm0 * 16 + 8 * half + r) * 64 + tn0 * 16 + (lane & 15)] = acc0[r];
    proj[(m0 + tm1 * 16 + 8 * half + r) * 64 + tn1 * 16 + (lane & 15)] = acc1[r];
  }
}

// Kernel 4: per-(token,head) src/trg scores
__global__ void k_gat1_scores(const float* __restrict__ proj, const float* __restrict__ asrc,
                              const float* __restrict__ atrg,
                              float* __restrict__ ssrc, float* __restrict__ strg) {
  const int idx = blockIdx.x * blockDim.x + threadIdx.x;  // m*8 + h, 131072
  const int m = idx >> 3, h = idx & 7;
  float s1 = 0.f, s2 = 0.f;
#pragma unroll
  for (int f = 0; f < F1; ++f) {
    const float p = proj[m * 64 + h * F1 + f];
    s1 += p * asrc[h * F1 + f];
    s2 += p * atrg[h * F1 + f];
  }
  ssrc[idx] = s1; strg[idx] = s2;
}

// Kernel 5: per-graph max m1[b] = max_h lrelu(max_s src + max_t trg)  (lrelu monotone)
__global__ void k_gat1_max(const float* __restrict__ ssrc, const float* __restrict__ strg,
                           float* __restrict__ m1) {
  const int b = blockIdx.x; const int tid = threadIdx.x;  // 64 threads
  __shared__ float red[64];
  float cand = -INFINITY;
  if (tid < NH1) {
    float ms = -INFINITY, mt = -INFINITY;
    for (int s = 0; s < Lq; ++s) {
      ms = fmaxf(ms, ssrc[(b * Lq + s) * NH1 + tid]);
      mt = fmaxf(mt, strg[(b * Lq + s) * NH1 + tid]);
    }
    const float v = ms + mt;
    cand = v > 0.f ? v : LRELU * v;
  }
  red[tid] = cand; __syncthreads();
  for (int off = 32; off >= 1; off >>= 1) {
    if (tid < off) red[tid] = fmaxf(red[tid], red[tid + off]);
    __syncthreads();
  }
  if (tid == 0) m1[b] = red[0];
}

// Kernel 6: GAT1 softmax-aggregate + bias + ELU. One block per (b, target t).
// Phase 1: 64 threads compute the 256x8 exp-weight matrix ONCE into LDS
// (32 exps/thread instead of 256) with per-head partial denominators.
// Phase 2: pure FMA stream over LDS weights + coalesced proj reads.
#define WS_S 9   // wsm stride (8 + 1 pad)
__global__ __launch_bounds__(64) void
k_gat1_agg(const float* __restrict__ proj, const float* __restrict__ ssrc,
           const float* __restrict__ strg, const float* __restrict__ m1,
           const float* __restrict__ g1b, float* __restrict__ h1) {
  __shared__ float wsm[Lq * WS_S];      // [s][h]
  __shared__ float dred[8][NH1];        // [s-chunk][h]
  const int b = blockIdx.x >> 8;
  const int t = blockIdx.x & 255;
  const int tid = threadIdx.x;          // 64
  const float m = m1[b];
  // phase 1: thread = (chunk = tid>>3, h2 = tid&7) covers s in [chunk*32, +32)
  const int h2 = tid & 7;
  const int sbase = (tid >> 3) * 32;
  const float trg2 = strg[(b * Lq + t) * NH1 + h2];
  float dpart = 0.f;
  for (int i = 0; i < 32; ++i) {
    const int s = sbase + i;
    float sc = ssrc[(b * Lq + s) * NH1 + h2] + trg2;
    sc = sc > 0.f ? sc : LRELU * sc;
    const float w = expf(sc - m);
    wsm[s * WS_S + h2] = w;
    dpart += w;
  }
  dred[tid >> 3][h2] = dpart;
  __syncthreads();
  // phase 2: thread = (h = tid>>3, f = tid&7)
  const int h = tid >> 3;
  float den = 0.f;
#pragma unroll
  for (int c = 0; c < 8; ++c) den += dred[c][h];
  float acc = 0.f;
  for (int s = 0; s < Lq; ++s)
    acc += wsm[s * WS_S + h] * proj[(b * Lq + s) * 64 + tid];
  float o = acc / (den + 1e-16f) + g1b[tid];
  o = o > 0.f ? o : expf(o) - 1.f;      // ELU
  h1[(b * Lq + t) * 64 + tid] = o;
}

// Kernel 7: GAT2 attention + context + masked softmax + max-pool. One block / batch.
__global__ void k_gat2_att(const float* __restrict__ h1, const float* __restrict__ g2W,
                           const float* __restrict__ g2src, const float* __restrict__ g2trg,
                           const float* __restrict__ ctx, const unsigned char* __restrict__ mask,
                           const float* __restrict__ hseq,
                           float* __restrict__ att_out, float* __restrict__ pooled) {
  const int b = blockIdx.x; const int tid = threadIdx.x;   // 256
  __shared__ float ps[Lq], pt[Lq], wctx[Lq], attL[Lq], red[Lq];
  float p = 0.f;
  for (int k = 0; k < 64; ++k) p += h1[(b * Lq + tid) * 64 + k] * g2W[k];
  const float s2 = p * g2src[0];
  const float t2 = p * g2trg[0];
  ps[tid] = s2; pt[tid] = t2;
  __syncthreads();
  red[tid] = s2; __syncthreads();
  for (int off = 128; off >= 1; off >>= 1) {
    if (tid < off) red[tid] = fmaxf(red[tid], red[tid + off]);
    __syncthreads();
  }
  const float maxs = red[0]; __syncthreads();
  red[tid] = t2; __syncthreads();
  for (int off = 128; off >= 1; off >>= 1) {
    if (tid < off) red[tid] = fmaxf(red[tid], red[tid + off]);
    __syncthreads();
  }
  const float maxt = red[0]; __syncthreads();
  float mm = maxs + maxt; mm = mm > 0.f ? mm : LRELU * mm;
  // denom over sources for target tid, then fold ctx/denom
  float den = 0.f;
  for (int s = 0; s < Lq; ++s) {
    float sc = ps[s] + t2;
    sc = sc > 0.f ? sc : LRELU * sc;
    den += expf(sc - mm);
  }
  wctx[tid] = ctx[tid] / (den + 1e-16f);
  __syncthreads();
  // att_raw[source = tid] = sum_t A2[s,t] * ctx[t]
  float a = 0.f;
  for (int t = 0; t < Lq; ++t) {
    float sc = s2 + pt[t];
    sc = sc > 0.f ? sc : LRELU * sc;
    a += expf(sc - mm) * wctx[t];
  }
  if (!mask[b * Lq + tid]) a = -INFINITY;
  // softmax over sources
  red[tid] = a; __syncthreads();
  for (int off = 128; off >= 1; off >>= 1) {
    if (tid < off) red[tid] = fmaxf(red[tid], red[tid + off]);
    __syncthreads();
  }
  const float amax = red[0]; __syncthreads();
  const float e = expf(a - amax);
  red[tid] = e; __syncthreads();
  for (int off = 128; off >= 1; off >>= 1) {
    if (tid < off) red[tid] = red[tid] + red[tid + off];
    __syncthreads();
  }
  const float av = e / red[0];
  attL[tid] = av;
  att_out[b * Lq + tid] = av;
  __syncthreads();
  // pooled[b][k=tid] = max_l hseq[b,l,k] * att[l]
  float pm = -INFINITY;
  for (int l = 0; l < Lq; ++l) pm = fmaxf(pm, hseq[(b * Lq + l) * TWOH + tid] * attL[l]);
  pooled[b * TWOH + tid] = pm;
}

// Kernel 8: hcl = relu(pooled @ lin_W.T + lin_b)
__global__ void k_hcl(const float* __restrict__ pooled, const float* __restrict__ linW,
                      const float* __restrict__ linb, float* __restrict__ hcl) {
  const int idx = blockIdx.x * blockDim.x + threadIdx.x;  // 4096: b*64+j
  const int b = idx >> 6, j = idx & 63;
  float s = linb[j];
  for (int k = 0; k < TWOH; ++k) s += pooled[b * TWOH + k] * linW[j * TWOH + k];
  hcl[idx] = s > 0.f ? s : 0.f;
}

// Kernel 9: logits = hcl @ out_W.T + out_b
__global__ void k_logits(const float* __restrict__ hcl, const float* __restrict__ outW,
                         const float* __restrict__ outb, float* __restrict__ logits) {
  const int idx = threadIdx.x;            // 128: b*2+c
  if (idx >= Bq * 2) return;
  const int b = idx >> 1, c = idx & 1;
  float s = outb[c];
  for (int j = 0; j < 64; ++j) s += hcl[b * 64 + j] * outW[c * 64 + j];
  logits[idx] = s;
}

// =====================================================================
extern "C" void kernel_launch(void* const* d_in, const int* in_sizes, int n_in,
                              void* d_out, int out_size, void* d_ws, size_t ws_size,
                              hipStream_t stream) {
  const int*           ids   = (const int*)d_in[0];
  const unsigned char* mask  = (const unsigned char*)d_in[1];
  const float* emb   = (const float*)d_in[2];
  const float* Wih_f = (const float*)d_in[3];
  const float* Whh_f = (const float*)d_in[4];
  const float* b_f   = (const float*)d_in[5];
  const float* Wih_b = (const float*)d_in[6];
  const float* Whh_b = (const float*)d_in[7];
  const float* b_b   = (const float*)d_in[8];
  const float* g1_W  = (const float*)d_in[9];
  const float* g1_sr = (const float*)d_in[10];
  const float* g1_tr = (const float*)d_in[11];
  const float* g1_b  = (const float*)d_in[12];
  const float* g2_W  = (const float*)d_in[13];
  const float* g2_sr = (const float*)d_in[14];
  const float* g2_tr = (const float*)d_in[15];
  /* d_in[16] = g2_b : unused (GAT2 output discarded, only A2 used) */
  const float* ctx   = (const float*)d_in[17];
  const float* lin_W = (const float*)d_in[18];
  const float* lin_b = (const float*)d_in[19];
  const float* out_W = (const float*)d_in[20];
  const float* out_b = (const float*)d_in[21];

  float* ws     = (float*)d_ws;
  float* xproj  = ws;                      // [16384][1024]        16,777,216 f
  float* hseq   = xproj + 16384 * 1024;    // [16384][256]          4,194,304 f
  float* proj1  = hseq  + 16384 * 256;     // [16384][64]           1,048,576 f
  float* ssrc   = proj1 + 16384 * 64;      // [16384][8]              131,072 f
  float* strg   = ssrc  + 16384 * 8;       // [16384][8]              131,072 f
  float* m1     = strg  + 16384 * 8;       // [64]
  float* h1     = m1    + 64;              // [16384][64]           1,048,576 f
  float* pooled = h1    + 16384 * 64;      // [64][256]                16,384 f
  float* hcl    = pooled + 64 * 256;       // [64][64]                  4,096 f
  // total ~93.4 MB of workspace

  float* logits = (float*)d_out;           // [64][2]
  float* att    = logits + Bq * 2;         // [64][256]

  k_embed_xproj<<<dim3(256, 16), 256, 0, stream>>>(ids, emb, Wih_f, Wih_b, b_f, b_b, xproj);
  k_lstm<<<2, 1024, SMEM_TOT, stream>>>(Whh_f, Whh_b, xproj, hseq);
  k_gat1_proj<<<256, 256, 0, stream>>>(hseq, g1_W, proj1);
  k_gat1_scores<<<512, 256, 0, stream>>>(proj1, g1_sr, g1_tr, ssrc, strg);
  k_gat1_max<<<64, 64, 0, stream>>>(ssrc, strg, m1);
  k_gat1_agg<<<16384, 64, 0, stream>>>(proj1, ssrc, strg, m1, g1_b, h1);
  k_gat2_att<<<64, 256, 0, stream>>>(h1, g2_W, g2_sr, g2_tr, ctx, mask, hseq, att, pooled);
  k_hcl<<<16, 256, 0, stream>>>(pooled, lin_W, lin_b, hcl);
  k_logits<<<1, 128, 0, stream>>>(hcl, out_W, out_b, logits);
}
